// Lorem_7997229105340
// MI455X (gfx1250) — matrix operand order repaired and checked by reading, not compiled, not run on hardware
//
#include <hip/hip_runtime.h>
#include <math.h>

// ---------------------------------------------------------------------------
// MPNN potential forward for MI455X (gfx1250, wave32, WMMA bf16 16x16x32).
// All GEMMs run on v_wmma_f32_16x16x32_bf16; edge tensor (P x 128) is fully
// fused (never hits HBM); segment sums exploit the 18-offset block structure
// of the pair list (no atomics). Ewald recip-space part is VALU sin/cos.
// Edge kernel: i-half of the feat GEMM hoisted out of the offset loop (LDS,
// f32) and volatile B-fragment loads inside the loop to stop LICM from
// hoisting ~800 VGPRs of weights into scratch spills.
// ---------------------------------------------------------------------------

#define DEV __device__ __forceinline__

typedef __attribute__((ext_vector_type(16))) __bf16       v16bf;
typedef __attribute__((ext_vector_type(8)))  float        v8f;
typedef __attribute__((ext_vector_type(4)))  unsigned int u32x4;

union Frag32B { u32x4 q[2]; v16bf bf; };

#define NA     32768     // atoms (32^3)
#define DFE    128       // feature dim D
#define NOFF   18        // neighbor offsets (|o|^2 in {1,2})
#define NKV    728       // Ewald k-vectors (9^3 - 1)
#define CUTR   5.0f
#define PI_F   3.14159265358979f
#define SIGMA_ 1.0f
#define LF1    1.31607401f   // 3^0.25
#define LF2    1.49534878f   // 5^0.25

__constant__ float BINOM31[32] = {
  1.f,31.f,465.f,4495.f,31465.f,169911.f,736281.f,2629575.f,
  7888725.f,20160075.f,44352165.f,84672315.f,141120525.f,206253075.f,
  265182525.f,300540195.f,300540195.f,265182525.f,206253075.f,141120525.f,
  84672315.f,44352165.f,20160075.f,7888725.f,2629575.f,736281.f,169911.f,
  31465.f,4495.f,465.f,31.f,1.f};

DEV unsigned short f2bf(float f){
  unsigned int u = __builtin_bit_cast(unsigned int, f);
  return (unsigned short)((u + 0x7FFFu + ((u >> 16) & 1u)) >> 16);
}

DEV float silu(float x){ return x / (1.f + expf(-x)); }

DEV v8f vzero(){
  v8f z;
#pragma unroll
  for (int i = 0; i < 8; ++i) z[i] = 0.f;
  return z;
}

DEV v16bf load_frag_2x16(const unsigned short* p0, const unsigned short* p1){
  Frag32B f;
  f.q[0] = *(const u32x4*)p0;
  f.q[1] = *(const u32x4*)p1;
  return f.bf;
}

// volatile variant: cannot be hoisted/CSE'd out of loops (anti-spill)
DEV v16bf load_frag_2x16_v(const unsigned short* p0, const unsigned short* p1){
  Frag32B f;
  f.q[0] = *(const volatile u32x4*)p0;
  f.q[1] = *(const volatile u32x4*)p1;
  return f.bf;
}

// A fragment (16x32 bf16, MxK): row = K-contiguous row of matrix row m=lane&15.
// lane holds 16 consecutive K at kh*16 with kh = lane>>4.
DEV v16bf load_afrag_row(const unsigned short* row, int kk, int lane){
  int base = kk*32 + (lane >> 4)*16;
  return load_frag_2x16(row + base, row + base + 8);
}

// B fragment (32x16 bf16, KxN) from transposed weights WT[n][k] (row len K):
// lane holds column n = nt*16 + lane&15, 16 consecutive K at kh*16.
DEV v16bf load_bfrag(const unsigned short* WT, int K, int nt, int kk, int lane){
  const unsigned short* r = WT + (size_t)(nt*16 + (lane & 15))*K + kk*32 + (lane >> 4)*16;
  return load_frag_2x16(r, r + 8);
}

// volatile B fragment with separate row stride (ldb) and column offset (kofs)
DEV v16bf load_bfrag_v(const unsigned short* WT, int ldb, int kofs, int nt, int kk, int lane){
  const unsigned short* r = WT + (size_t)(nt*16 + (lane & 15))*ldb + kofs + kk*32 + (lane >> 4)*16;
  return load_frag_2x16_v(r, r + 8);
}

DEV v8f wmma_bf16(v16bf a, v16bf b, v8f c){
  return __builtin_amdgcn_wmma_f32_16x16x32_bf16(false, a, false, b, (short)0, c, false, false);
}

template<int K32>
DEV v8f gemm_row(const unsigned short* arow, const unsigned short* WT, int nt, int lane, v8f acc){
#pragma unroll
  for (int kk = 0; kk < K32; ++kk)
    acc = wmma_bf16(load_afrag_row(arow, kk, lane), load_bfrag(WT, K32*32, nt, kk, lane), acc);
  return acc;
}

// D fragment (16x16 f32): reg r holds M = r + 8*(lane>>4), N = nt*16 + lane&15
DEV void store_d_bf(unsigned short* buf, int stride, int nt, int lane, v8f d){
  int n = nt*16 + (lane & 15);
  int kh = lane >> 4;
#pragma unroll
  for (int r = 0; r < 8; ++r)
    buf[(r + 8*kh)*stride + n] = f2bf(d[r]);
}

// ---------------------------------------------------------------------------
// Prep kernels
// ---------------------------------------------------------------------------
__global__ void mpnn_prep_wt(const float* __restrict__ W, unsigned short* __restrict__ WT,
                             int K, int Nout, int Kpad, int Npad){
  int idx = blockIdx.x*blockDim.x + threadIdx.x;
  int tot = Kpad*Npad;
  if (idx >= tot) return;
  int n = idx / Kpad, k = idx % Kpad;
  float v = (k < K && n < Nout) ? W[(size_t)k*Nout + n] : 0.f;
  WT[idx] = f2bf(v);
}

__global__ void mpnn_species(const float* __restrict__ embed, const int* __restrict__ Z,
                             const float* __restrict__ amask, unsigned short* __restrict__ spbf){
  int idx = blockIdx.x*blockDim.x + threadIdx.x;
  if (idx >= NA*DFE) return;
  int n = idx >> 7, d = idx & 127;
  spbf[idx] = f2bf(embed[Z[n]*DFE + d] * amask[n]);
}

__global__ void mpnn_ewald_prep(const float* __restrict__ cell, float* __restrict__ kt){
  int t = blockIdx.x*blockDim.x + threadIdx.x;
  if (t >= NKV) return;
  float c00=cell[0],c01=cell[1],c02=cell[2];
  float c10=cell[3],c11=cell[4],c12=cell[5];
  float c20=cell[6],c21=cell[7],c22=cell[8];
  float det = c00*(c11*c22-c12*c21) - c01*(c10*c22-c12*c20) + c02*(c10*c21-c11*c20);
  float V = fabsf(det);
  float idet = 1.f/det;
  const float TP = 6.28318530718f;
  // B = 2*pi * inv(cell)^T  =>  B[i][j] = 2*pi*cof[i][j]/det
  float B00 =  (c11*c22-c12*c21)*idet*TP;
  float B01 = -(c10*c22-c12*c20)*idet*TP;
  float B02 =  (c10*c21-c11*c20)*idet*TP;
  float B10 = -(c01*c22-c02*c21)*idet*TP;
  float B11 =  (c00*c22-c02*c20)*idet*TP;
  float B12 = -(c00*c21-c01*c20)*idet*TP;
  float B20 =  (c01*c12-c02*c11)*idet*TP;
  float B21 = -(c00*c12-c02*c10)*idet*TP;
  float B22 =  (c00*c11-c01*c10)*idet*TP;
  int tt = (t < 364) ? t : t + 1;           // skip (0,0,0)
  float nx = (float)(tt/81) - 4.f;
  float ny = (float)((tt/9)%9) - 4.f;
  float nz = (float)(tt%9) - 4.f;
  float kx = nx*B00 + ny*B10 + nz*B20;
  float ky = nx*B01 + ny*B11 + nz*B21;
  float kz = nx*B02 + ny*B12 + nz*B22;
  float k2 = kx*kx + ky*ky + kz*kz;
  float coef = (4.f*PI_F/V) * expf(-0.5f*SIGMA_*SIGMA_*k2) / k2;
  kt[t*4+0]=kx; kt[t*4+1]=ky; kt[t*4+2]=kz; kt[t*4+3]=coef;
}

// ---------------------------------------------------------------------------
// Edge kernel: one wave owns 16 atoms and all their 18 edges (fused).
// 2 waves per 64-thread block (LDS: 2 x 18.2KB).
// ---------------------------------------------------------------------------
struct EdgeSmem {
  int   jidx[16];
  float cut[16];
  float Y[16][9];
  alignas(16) unsigned short rad[16][32];
  alignas(16) unsigned short edges[16][DFE];
  float sph[16][72];
  float ebase[16][DFE];    // sp_i @ Wf_top + bias  (offset-invariant half)
};

__global__ void __launch_bounds__(64) mpnn_edges(
  const float* __restrict__ pos, const float* __restrict__ cell,
  const float* __restrict__ shifts, const float* __restrict__ pmask,
  const float* __restrict__ amask, const int* __restrict__ others,
  const unsigned short* __restrict__ spbf,
  const unsigned short* __restrict__ WfT, const float* __restrict__ bf_bias,
  const unsigned short* __restrict__ WrT, const unsigned short* __restrict__ msgT,
  const unsigned short* __restrict__ sphT,
  float* __restrict__ updws, float* __restrict__ sphraw)
{
  __shared__ EdgeSmem sm[2];
  const int wave = threadIdx.x >> 5, lane = threadIdx.x & 31;
  EdgeSmem& S = sm[wave];
  const int tile = (blockIdx.x*2 + wave)*16;
  const int kh = lane >> 4, ml = lane & 15;

  float mk[8];
#pragma unroll
  for (int r = 0; r < 8; ++r) mk[r] = amask[tile + r + 8*kh];

  for (int t = lane; t < 16*72; t += 32) (&S.sph[0][0])[t] = 0.f;

  // Hoisted i-half: ebase = sp_i @ Wf[k=0..127] + bias (same for all offsets)
  {
    const unsigned short* arow = spbf + (size_t)(tile + ml)*DFE;
    const int n = lane & 15;   // column within tile of 16
#pragma unroll
    for (int nt = 0; nt < 8; ++nt){
      v8f acc = vzero();
#pragma unroll
      for (int kk = 0; kk < 4; ++kk)
        acc = wmma_bf16(load_afrag_row(arow, kk, lane),
                        load_bfrag(WfT + 0, 256, nt, kk, lane), acc);
      float bn = bf_bias[nt*16 + n];
#pragma unroll
      for (int r = 0; r < 8; ++r)
        S.ebase[r + 8*kh][nt*16 + n] = acc[r] + bn;
    }
  }

  v8f upd[8];
#pragma unroll
  for (int nt = 0; nt < 8; ++nt)
#pragma unroll
    for (int i = 0; i < 8; ++i) upd[nt][i] = 0.f;

  for (int off = 0; off < NOFF; ++off){
    if (lane < 16){
      const int a = tile + lane;
      const int p = off*NA + a;
      const int j = others[p];
      S.jidx[lane] = j;
      float wx = shifts[(size_t)p*3+0], wy = shifts[(size_t)p*3+1], wz = shifts[(size_t)p*3+2];
      float sx = wx*cell[0] + wy*cell[3] + wz*cell[6];
      float sy = wx*cell[1] + wy*cell[4] + wz*cell[7];
      float sz = wx*cell[2] + wy*cell[5] + wz*cell[8];
      float rx = pos[(size_t)j*3+0] - pos[(size_t)a*3+0] + sx;
      float ry = pos[(size_t)j*3+1] - pos[(size_t)a*3+1] + sy;
      float rz = pos[(size_t)j*3+2] - pos[(size_t)a*3+2] + sz;
      float r = sqrtf(rx*rx + ry*ry + rz*rz + 1e-12f);
      float inv = 1.f/r;
      float x = rx*inv, y = ry*inv, z = rz*inv;
      float cw = (r < CUTR) ? 0.5f*(cosf(PI_F*r*(1.f/CUTR)) + 1.f) : 0.f;
      S.cut[lane] = cw * pmask[p];
      const float SQ3 = 1.73205080757f;
      S.Y[lane][0] = 1.f; S.Y[lane][1] = x; S.Y[lane][2] = y; S.Y[lane][3] = z;
      S.Y[lane][4] = SQ3*x*y; S.Y[lane][5] = SQ3*y*z;
      S.Y[lane][6] = 0.5f*(3.f*z*z - 1.f);
      S.Y[lane][7] = SQ3*x*z; S.Y[lane][8] = 0.5f*SQ3*(x*x - y*y);
      float tt  = fminf(fmaxf(r*(1.f/CUTR), 1e-6f), 1.f - 1e-6f);
      float omt = 1.f - tt, iom = 1.f/omt;
      float pw = 1.f;
#pragma unroll
      for (int q = 0; q < 31; ++q) pw *= omt;   // (1-t)^31
      float ak = 1.f;
#pragma unroll
      for (int k = 0; k < 32; ++k){
        S.rad[lane][k] = f2bf(BINOM31[k]*ak*pw); // C(31,k) t^k (1-t)^(31-k)
        ak *= tt; pw *= iom;
      }
    }
    __syncthreads();

    const unsigned short* arowj = spbf + (size_t)S.jidx[ml]*DFE;
    const unsigned short* radrow = &S.rad[ml][0];

    // edges_scalar = (ebase + sp_j @ Wf_bot) * (radial @ Wr) * cutoff -> bf16
#pragma unroll
    for (int nt = 0; nt < 8; ++nt){
      v8f rf = wmma_bf16(load_afrag_row(radrow, 0, lane),
                         load_bfrag_v(WrT, 32, 0, nt, 0, lane), vzero());
      v8f acc = vzero();
#pragma unroll
      for (int kk = 0; kk < 4; ++kk)
        acc = wmma_bf16(load_afrag_row(arowj, kk, lane),
                        load_bfrag_v(WfT, 256, 128, nt, kk, lane), acc);
#pragma unroll
      for (int r = 0; r < 8; ++r){
        int m = r + 8*kh;
        float v = (acc[r] + S.ebase[m][nt*16 + ml]) * rf[r] * S.cut[m];
        S.edges[m][nt*16 + ml] = f2bf(v);
      }
    }
    __syncthreads();

    const unsigned short* erow = &S.edges[ml][0];

    // message GEMM: upd[i] += edges @ msg_W  (register-resident across offsets)
#pragma unroll
    for (int nt = 0; nt < 8; ++nt)
#pragma unroll
      for (int kk = 0; kk < 4; ++kk)
        upd[nt] = wmma_bf16(load_afrag_row(erow, kk, lane),
                            load_bfrag_v(msgT, 128, 0, nt, kk, lane), upd[nt]);

    // sph GEMM: coeff = edges @ sph_W (24 cols, padded 32); scatter *Y into LDS
#pragma unroll
    for (int nt2 = 0; nt2 < 2; ++nt2){
      v8f cf = vzero();
#pragma unroll
      for (int kk = 0; kk < 4; ++kk)
        cf = wmma_bf16(load_afrag_row(erow, kk, lane),
                       load_bfrag_v(sphT, 128, 0, nt2, kk, lane), cf);
      int c = nt2*16 + ml;
      if (c < 24){
        int dg = c >> 3, s = c & 7;
        int lm0 = (dg == 0) ? 0 : ((dg == 1) ? 1 : 4);
        int nlm = (dg == 0) ? 1 : ((dg == 1) ? 3 : 5);
#pragma unroll
        for (int r = 0; r < 8; ++r){
          int m = r + 8*kh;
          float cv = cf[r];
          for (int q = 0; q < nlm; ++q){
            int lm = lm0 + q;
            S.sph[m][lm*8 + s] += cv * S.Y[m][lm];  // unique owner lane: race-free
          }
        }
      }
    }
    __syncthreads();
  }

  // write segment-summed message update and raw nodes_sph
#pragma unroll
  for (int nt = 0; nt < 8; ++nt){
    int n = nt*16 + ml;
#pragma unroll
    for (int r = 0; r < 8; ++r){
      int a = tile + r + 8*kh;
      updws[(size_t)a*DFE + n] = upd[nt][r] * mk[r];
    }
  }
  for (int t = lane; t < 16*72; t += 32){
    int m = t/72, c = t%72;
    int a = tile + m;
    sphraw[(size_t)a*72 + c] = S.sph[m][c] * amask[a];
  }
}

// ---------------------------------------------------------------------------
// Node kernel 1: init, u1, sph-transform/norms, u2, m1 energy, charge heads.
// ---------------------------------------------------------------------------
__global__ void __launch_bounds__(128) mpnn_node1(
  const unsigned short* __restrict__ spbf,
  const unsigned short* __restrict__ initT, const float* __restrict__ init_b,
  const float* __restrict__ updws,
  const unsigned short* __restrict__ u1W1T, const float* __restrict__ u1b1,
  const unsigned short* __restrict__ u1W2T, const float* __restrict__ u1b2,
  const float* __restrict__ sphraw, const float* __restrict__ tdW,
  const unsigned short* __restrict__ u2W1T, const float* __restrict__ u2b1,
  const unsigned short* __restrict__ u2W2T, const float* __restrict__ u2b2,
  const unsigned short* __restrict__ m1W1T, const float* __restrict__ m1b1,
  const unsigned short* __restrict__ m1W2T, const float* __restrict__ m1b2,
  const unsigned short* __restrict__ m1W3T, const float* __restrict__ m1b3,
  const unsigned short* __restrict__ qW1T, const float* __restrict__ qb1,
  const unsigned short* __restrict__ qW2T, const float* __restrict__ qb2,
  const float* __restrict__ qsW, const float* __restrict__ amask,
  float* __restrict__ nodesws, float* __restrict__ sph2ws,
  float* __restrict__ chargesws, float* __restrict__ e1ws)
{
  __shared__ alignas(16) unsigned short buf0[4][16*256];
  __shared__ alignas(16) unsigned short buf1[4][16*128];
  __shared__ alignas(16) unsigned short buf2[4][16*32];
  const int wave = threadIdx.x >> 5, lane = threadIdx.x & 31;
  const int tile = (blockIdx.x*4 + wave)*16;
  const int kh = lane >> 4, ml = lane & 15;
  unsigned short* B0 = buf0[wave];
  unsigned short* B1 = buf1[wave];
  unsigned short* B2 = buf2[wave];

  float mk[8];
#pragma unroll
  for (int r = 0; r < 8; ++r) mk[r] = amask[tile + r + 8*kh];

  v8f nd[8];

  // 1) nodes = (species @ init_W + b) * mask
  {
    const unsigned short* arow = spbf + (size_t)(tile + ml)*DFE;
#pragma unroll
    for (int nt = 0; nt < 8; ++nt){
      v8f acc = gemm_row<4>(arow, initT, nt, lane, vzero());
      float bn = init_b[nt*16 + ml];
#pragma unroll
      for (int r = 0; r < 8; ++r) nd[nt][r] = (acc[r] + bn) * mk[r];
    }
  }
  // 2) stage message update (bf16) into B0
  for (int t = lane; t < 16*DFE; t += 32){
    int m = t >> 7, n = t & 127;
    B0[m*256 + n] = f2bf(updws[(size_t)(tile+m)*DFE + n]);
  }
  __syncthreads();
  // 3) u1: h = silu(upd @ W1 + b1); nodes = (nodes + h @ W2 + b2) * mask
  {
    const unsigned short* ar = B0 + ml*256;
#pragma unroll
    for (int nt = 0; nt < 8; ++nt){
      v8f h = gemm_row<4>(ar, u1W1T, nt, lane, vzero());
      float bn = u1b1[nt*16 + ml];
#pragma unroll
      for (int r = 0; r < 8; ++r) h[r] = silu(h[r] + bn);
      store_d_bf(B1, 128, nt, lane, h);
    }
  }
  __syncthreads();
  {
    const unsigned short* ar = B1 + ml*128;
#pragma unroll
    for (int nt = 0; nt < 8; ++nt){
      v8f d = gemm_row<4>(ar, u1W2T, nt, lane, vzero());
      float bn = u1b2[nt*16 + ml];
#pragma unroll
      for (int r = 0; r < 8; ++r) nd[nt][r] = (nd[nt][r] + d[r] + bn) * mk[r];
    }
  }
  __syncthreads();
  // 5) sph einsum with td_W + per-l norms -> upd2 (bf16 in B2)
  for (int u = 0; u < 4; ++u){
    int idx = lane + 32*u;          // 128 (m,g) items
    int m = idx >> 3, g = idx & 7;
    int a = tile + m;
    float o[9];
#pragma unroll
    for (int lm = 0; lm < 9; ++lm){
      int dg = (lm == 0) ? 0 : ((lm < 4) ? 1 : 2);
      float s = 0.f;
#pragma unroll
      for (int f = 0; f < 8; ++f)
        s += sphraw[(size_t)a*72 + lm*8 + f] * tdW[(dg*8+f)*8 + g];
      o[lm] = s;
      sph2ws[(size_t)a*72 + lm*8 + g] = s;
    }
    float n0 = sqrtf(o[0]*o[0] + 1e-12f);
    float n1 = sqrtf(o[1]*o[1] + o[2]*o[2] + o[3]*o[3] + 1e-12f);
    float n2 = sqrtf(o[4]*o[4] + o[5]*o[5] + o[6]*o[6] + o[7]*o[7] + o[8]*o[8] + 1e-12f);
    B2[m*32 + 0  + g] = f2bf(n0);
    B2[m*32 + 8  + g] = f2bf(n1*LF1);
    B2[m*32 + 16 + g] = f2bf(n2*LF2);
  }
  for (int t = lane; t < 16*8; t += 32){ int m = t >> 3; B2[m*32 + 24 + (t&7)] = 0; }
  __syncthreads();
  // 6) u2 update
  {
    const unsigned short* ar = B2 + ml*32;
#pragma unroll
    for (int nt = 0; nt < 8; ++nt){
      v8f h = gemm_row<1>(ar, u2W1T, nt, lane, vzero());
      float bn = u2b1[nt*16 + ml];
#pragma unroll
      for (int r = 0; r < 8; ++r) h[r] = silu(h[r] + bn);
      store_d_bf(B1, 128, nt, lane, h);
    }
  }
  __syncthreads();
  {
    const unsigned short* ar = B1 + ml*128;
#pragma unroll
    for (int nt = 0; nt < 8; ++nt){
      v8f d = gemm_row<4>(ar, u2W2T, nt, lane, vzero());
      float bn = u2b2[nt*16 + ml];
#pragma unroll
      for (int r = 0; r < 8; ++r) nd[nt][r] = (nd[nt][r] + d[r] + bn) * mk[r];
    }
  }
  __syncthreads();
  // 7) persist nodes f32 + stage bf16 into B1
#pragma unroll
  for (int nt = 0; nt < 8; ++nt){
    int n = nt*16 + ml;
#pragma unroll
    for (int r = 0; r < 8; ++r)
      nodesws[(size_t)(tile + r + 8*kh)*DFE + n] = nd[nt][r];
    store_d_bf(B1, 128, nt, lane, nd[nt]);
  }
  __syncthreads();
  // 8) charge head: qh = silu(nodes @ qW1 + b1) (256 wide); scalar_q = qh @ qW2 + b2
  {
    const unsigned short* ar = B1 + ml*128;
#pragma unroll
    for (int nt = 0; nt < 16; ++nt){
      v8f h = gemm_row<4>(ar, qW1T, nt, lane, vzero());
      float bn = qb1[nt*16 + ml];
#pragma unroll
      for (int r = 0; r < 8; ++r) h[r] = silu(h[r] + bn);
      store_d_bf(B0, 256, nt, lane, h);
    }
  }
  __syncthreads();
  {
    const unsigned short* ar = B0 + ml*256;
    v8f sq = gemm_row<8>(ar, qW2T, 0, lane, vzero());
    if (ml == 0){
#pragma unroll
      for (int r = 0; r < 8; ++r)
        chargesws[(size_t)(tile + r + 8*kh)*10 + 0] = (sq[r] + qb2[0]) * mk[r];
    }
  }
  // 9) sph charges
  for (int t = lane; t < 16*9; t += 32){
    int m = t/9, lm = t%9;
    int a = tile + m;
    int dg = (lm == 0) ? 0 : ((lm < 4) ? 1 : 2);
    float s = 0.f;
#pragma unroll
    for (int f = 0; f < 8; ++f) s += sph2ws[(size_t)a*72 + lm*8 + f] * qsW[dg*8 + f];
    chargesws[(size_t)a*10 + 1 + lm] = s * amask[a];
  }
  __syncthreads();
  // 10) m1 energy MLP
  {
    const unsigned short* ar = B1 + ml*128;  // nodes bf16
#pragma unroll
    for (int nt = 0; nt < 8; ++nt){
      v8f h = gemm_row<4>(ar, m1W1T, nt, lane, vzero());
      float bn = m1b1[nt*16 + ml];
#pragma unroll
      for (int r = 0; r < 8; ++r) h[r] = silu(h[r] + bn);
      store_d_bf(B0, 128, nt, lane, h);
    }
  }
  __syncthreads();
  {
    const unsigned short* ar = B0 + ml*128;
#pragma unroll
    for (int nt = 0; nt < 8; ++nt){
      v8f h = gemm_row<4>(ar, m1W2T, nt, lane, vzero());
      float bn = m1b2[nt*16 + ml];
#pragma unroll
      for (int r = 0; r < 8; ++r) h[r] = silu(h[r] + bn);
      store_d_bf(B1, 128, nt, lane, h);
    }
  }
  __syncthreads();
  {
    const unsigned short* ar = B1 + ml*128;
    v8f e = gemm_row<4>(ar, m1W3T, 0, lane, vzero());
    if (ml == 0){
#pragma unroll
      for (int r = 0; r < 8; ++r)
        e1ws[tile + r + 8*kh] = (e[r] + m1b3[0]) * mk[r];
    }
  }
}

// ---------------------------------------------------------------------------
// Ewald: structure factors (wave per k-vector) and per-atom potentials.
// ---------------------------------------------------------------------------
__global__ void __launch_bounds__(128) mpnn_sfact(
  const float* __restrict__ pos, const float* __restrict__ charges,
  const float* __restrict__ kt, float* __restrict__ ScC, float* __restrict__ SsC)
{
  const int wave = threadIdx.x >> 5, lane = threadIdx.x & 31;
  const int kidx = blockIdx.x*4 + wave;
  const float kx = kt[kidx*4+0], ky = kt[kidx*4+1], kz = kt[kidx*4+2], co = kt[kidx*4+3];
  float ac[10], as[10];
#pragma unroll
  for (int o = 0; o < 10; ++o){ ac[o] = 0.f; as[o] = 0.f; }
  for (int a = lane; a < NA; a += 32){
    float kr = kx*pos[(size_t)a*3+0] + ky*pos[(size_t)a*3+1] + kz*pos[(size_t)a*3+2];
    float s = sinf(kr), c = cosf(kr);
#pragma unroll
    for (int o = 0; o < 10; ++o){
      float q = charges[(size_t)a*10 + o];
      ac[o] += c*q; as[o] += s*q;
    }
  }
#pragma unroll
  for (int o = 0; o < 10; ++o){
    for (int d = 16; d > 0; d >>= 1){
      ac[o] += __shfl_down(ac[o], d, 32);
      as[o] += __shfl_down(as[o], d, 32);
    }
  }
  if (lane == 0){
#pragma unroll
    for (int o = 0; o < 10; ++o){
      ScC[kidx*10 + o] = co * ac[o];
      SsC[kidx*10 + o] = co * as[o];
    }
  }
}

__global__ void __launch_bounds__(256) mpnn_pot(
  const float* __restrict__ pos, const float* __restrict__ charges,
  const float* __restrict__ kt, const float* __restrict__ ScC,
  const float* __restrict__ SsC, const float* __restrict__ amask,
  float* __restrict__ potws)
{
  int a = blockIdx.x*256 + threadIdx.x;
  float px = pos[(size_t)a*3+0], py = pos[(size_t)a*3+1], pz = pos[(size_t)a*3+2];
  float acc[10];
#pragma unroll
  for (int o = 0; o < 10; ++o) acc[o] = 0.f;
  for (int k = 0; k < NKV; ++k){
    float kr = kt[k*4+0]*px + kt[k*4+1]*py + kt[k*4+2]*pz;
    float s = sinf(kr), c = cosf(kr);
#pragma unroll
    for (int o = 0; o < 10; ++o)
      acc[o] += c*ScC[k*10 + o] + s*SsC[k*10 + o];
  }
  const float selfc = 0.79788456080f / SIGMA_;   // sqrt(2/pi)/sigma
  float m = amask[a];
#pragma unroll
  for (int o = 0; o < 10; ++o)
    potws[(size_t)a*10 + o] = (acc[o] - charges[(size_t)a*10 + o]*selfc) * m;
}

// ---------------------------------------------------------------------------
// Node kernel 2: u3 update + m2 energy MLP -> final per-atom energy.
// ---------------------------------------------------------------------------
__global__ void __launch_bounds__(128) mpnn_node2(
  const float* __restrict__ nodesws, const float* __restrict__ sph2ws,
  const float* __restrict__ potws, const float* __restrict__ potW,
  const unsigned short* __restrict__ u3W1T, const float* __restrict__ u3b1,
  const unsigned short* __restrict__ u3W2T, const float* __restrict__ u3b2,
  const unsigned short* __restrict__ m2W1T, const float* __restrict__ m2b1,
  const unsigned short* __restrict__ m2W2T, const float* __restrict__ m2b2,
  const unsigned short* __restrict__ m2W3T, const float* __restrict__ m2b3,
  const float* __restrict__ amask, const float* __restrict__ e1ws,
  float* __restrict__ outE)
{
  __shared__ alignas(16) unsigned short buf0[4][16*128];
  __shared__ alignas(16) unsigned short buf1[4][16*128];
  __shared__ alignas(16) unsigned short buf2[4][16*32];
  const int wave = threadIdx.x >> 5, lane = threadIdx.x & 31;
  const int tile = (blockIdx.x*4 + wave)*16;
  const int kh = lane >> 4, ml = lane & 15;
  unsigned short* B0 = buf0[wave];
  unsigned short* B1 = buf1[wave];
  unsigned short* B2 = buf2[wave];

  float mk[8];
#pragma unroll
  for (int r = 0; r < 8; ++r) mk[r] = amask[tile + r + 8*kh];

  // upd3 = [scalar_pot, norms(sph_pot * nodes_sph) * LF]  (25 ch, padded 32)
  for (int t = lane; t < 16*8; t += 32){
    int m = t >> 3, g = t & 7;
    int a = tile + m;
    float su[9];
#pragma unroll
    for (int lm = 0; lm < 9; ++lm)
      su[lm] = potws[(size_t)a*10 + 1 + lm] * potW[g] * sph2ws[(size_t)a*72 + lm*8 + g];
    float n0 = sqrtf(su[0]*su[0] + 1e-12f);
    float n1 = sqrtf(su[1]*su[1] + su[2]*su[2] + su[3]*su[3] + 1e-12f);
    float n2 = sqrtf(su[4]*su[4] + su[5]*su[5] + su[6]*su[6] + su[7]*su[7] + su[8]*su[8] + 1e-12f);
    B2[m*32 + 1 + 0  + g] = f2bf(n0);
    B2[m*32 + 1 + 8  + g] = f2bf(n1*LF1);
    B2[m*32 + 1 + 16 + g] = f2bf(n2*LF2);
    if (g == 0) B2[m*32 + 0] = f2bf(potws[(size_t)a*10 + 0]);
  }
  for (int t = lane; t < 16*7; t += 32){ int m = t/7; B2[m*32 + 25 + t%7] = 0; }
  __syncthreads();

  // u3: h = silu(upd3 @ W1 + b1)
  {
    const unsigned short* ar = B2 + ml*32;
#pragma unroll
    for (int nt = 0; nt < 8; ++nt){
      v8f h = gemm_row<1>(ar, u3W1T, nt, lane, vzero());
      float bn = u3b1[nt*16 + ml];
#pragma unroll
      for (int r = 0; r < 8; ++r) h[r] = silu(h[r] + bn);
      store_d_bf(B1, 128, nt, lane, h);
    }
  }
  __syncthreads();
  v8f nd[8];
  {
    const unsigned short* ar = B1 + ml*128;
#pragma unroll
    for (int nt = 0; nt < 8; ++nt){
      v8f d = gemm_row<4>(ar, u3W2T, nt, lane, vzero());
      float bn = u3b2[nt*16 + ml];
      int n = nt*16 + ml;
#pragma unroll
      for (int r = 0; r < 8; ++r){
        float old = nodesws[(size_t)(tile + r + 8*kh)*DFE + n];
        nd[nt][r] = (old + d[r] + bn) * mk[r];
      }
    }
  }
  __syncthreads();
#pragma unroll
  for (int nt = 0; nt < 8; ++nt) store_d_bf(B0, 128, nt, lane, nd[nt]);
  __syncthreads();
  // m2 MLP
  {
    const unsigned short* ar = B0 + ml*128;
#pragma unroll
    for (int nt = 0; nt < 8; ++nt){
      v8f h = gemm_row<4>(ar, m2W1T, nt, lane, vzero());
      float bn = m2b1[nt*16 + ml];
#pragma unroll
      for (int r = 0; r < 8; ++r) h[r] = silu(h[r] + bn);
      store_d_bf(B1, 128, nt, lane, h);
    }
  }
  __syncthreads();
  {
    const unsigned short* ar = B1 + ml*128;
#pragma unroll
    for (int nt = 0; nt < 8; ++nt){
      v8f h = gemm_row<4>(ar, m2W2T, nt, lane, vzero());
      float bn = m2b2[nt*16 + ml];
#pragma unroll
      for (int r = 0; r < 8; ++r) h[r] = silu(h[r] + bn);
      store_d_bf(B0, 128, nt, lane, h);
    }
  }
  __syncthreads();
  {
    const unsigned short* ar = B0 + ml*128;
    v8f e = gemm_row<4>(ar, m2W3T, 0, lane, vzero());
    if (ml == 0){
#pragma unroll
      for (int r = 0; r < 8; ++r){
        int a = tile + r + 8*kh;
        outE[a] = e1ws[a] + (e[r] + m2b3[0]) * mk[r];
      }
    }
  }
}

// ---------------------------------------------------------------------------
// Host launcher
// ---------------------------------------------------------------------------
extern "C" void kernel_launch(void* const* d_in, const int* in_sizes, int n_in,
                              void* d_out, int out_size, void* d_ws, size_t ws_size,
                              hipStream_t stream)
{
  (void)in_sizes; (void)n_in; (void)out_size; (void)ws_size;
  const float* positions = (const float*)d_in[0];
  const float* cell      = (const float*)d_in[1];
  const float* shifts    = (const float*)d_in[2];
  const float* pmask     = (const float*)d_in[3];
  const float* amask     = (const float*)d_in[4];
  const float* W_embed = (const float*)d_in[5];
  const float* W_rc1Wf = (const float*)d_in[6];
  const float* b_rc1   = (const float*)d_in[7];
  const float* W_rc1Wr = (const float*)d_in[8];
  const float* W_init  = (const float*)d_in[9];
  const float* b_init  = (const float*)d_in[10];
  const float* W_msg   = (const float*)d_in[11];
  const float* W_u1W1  = (const float*)d_in[12];
  const float* b_u1b1  = (const float*)d_in[13];
  const float* W_u1W2  = (const float*)d_in[14];
  const float* b_u1b2  = (const float*)d_in[15];
  const float* W_sph   = (const float*)d_in[16];
  const float* W_td    = (const float*)d_in[17];
  const float* W_u2W1  = (const float*)d_in[18];
  const float* b_u2b1  = (const float*)d_in[19];
  const float* W_u2W2  = (const float*)d_in[20];
  const float* b_u2b2  = (const float*)d_in[21];
  const float* W_m1W1  = (const float*)d_in[22];
  const float* b_m1b1  = (const float*)d_in[23];
  const float* W_m1W2  = (const float*)d_in[24];
  const float* b_m1b2  = (const float*)d_in[25];
  const float* W_m1W3  = (const float*)d_in[26];
  const float* b_m1b3  = (const float*)d_in[27];
  const float* W_qW1   = (const float*)d_in[28];
  const float* b_qb1   = (const float*)d_in[29];
  const float* W_qW2   = (const float*)d_in[30];
  const float* b_qb2   = (const float*)d_in[31];
  const float* W_qs    = (const float*)d_in[32];
  const float* W_pot   = (const float*)d_in[33];
  const float* W_u3W1  = (const float*)d_in[34];
  const float* b_u3b1  = (const float*)d_in[35];
  const float* W_u3W2  = (const float*)d_in[36];
  const float* b_u3b2  = (const float*)d_in[37];
  const float* W_m2W1  = (const float*)d_in[38];
  const float* b_m2b1  = (const float*)d_in[39];
  const float* W_m2W2  = (const float*)d_in[40];
  const float* b_m2b2  = (const float*)d_in[41];
  const float* W_m2W3  = (const float*)d_in[42];
  const float* b_m2b3  = (const float*)d_in[43];
  const int* Zi     = (const int*)d_in[44];
  const int* others = (const int*)d_in[46];

  char* ws = (char*)d_ws;
  size_t cur = 0;
  auto take = [&](size_t bytes)->size_t {
    size_t o = cur; cur += (bytes + 255) & ~(size_t)255; return o;
  };
  size_t o_spbf   = take((size_t)NA*DFE*2);
  size_t o_upd    = take((size_t)NA*DFE*4);
  size_t o_sphraw = take((size_t)NA*72*4);
  size_t o_sph2   = take((size_t)NA*72*4);
  size_t o_nodes  = take((size_t)NA*DFE*4);
  size_t o_chg    = take((size_t)NA*10*4);
  size_t o_potw   = take((size_t)NA*10*4);
  size_t o_e1     = take((size_t)NA*4);
  size_t o_kt     = take((size_t)NKV*4*4);
  size_t o_ScC    = take((size_t)NKV*10*4);
  size_t o_SsC    = take((size_t)NKV*10*4);
  size_t o_WfT    = take((size_t)128*256*2);
  size_t o_WrT    = take((size_t)128*32*2);
  size_t o_initT  = take((size_t)128*128*2);
  size_t o_msgT   = take((size_t)128*128*2);
  size_t o_sphT   = take((size_t)32*128*2);
  size_t o_u1W1T  = take((size_t)128*128*2);
  size_t o_u1W2T  = take((size_t)128*128*2);
  size_t o_u2W1T  = take((size_t)128*32*2);
  size_t o_u2W2T  = take((size_t)128*128*2);
  size_t o_m1W1T  = take((size_t)128*128*2);
  size_t o_m1W2T  = take((size_t)128*128*2);
  size_t o_m1W3T  = take((size_t)16*128*2);
  size_t o_qW1T   = take((size_t)256*128*2);
  size_t o_qW2T   = take((size_t)16*256*2);
  size_t o_u3W1T  = take((size_t)128*32*2);
  size_t o_u3W2T  = take((size_t)128*128*2);
  size_t o_m2W1T  = take((size_t)128*128*2);
  size_t o_m2W2T  = take((size_t)128*128*2);
  size_t o_m2W3T  = take((size_t)16*128*2);

  auto US = [&](size_t off)->unsigned short* { return (unsigned short*)(ws + off); };
  auto F  = [&](size_t off)->float*          { return (float*)(ws + off); };

  auto WT = [&](const float* W, size_t off, int K, int Nout, int Kpad, int Npad){
    int tot = Kpad*Npad;
    mpnn_prep_wt<<<(tot + 255)/256, 256, 0, stream>>>(W, US(off), K, Nout, Kpad, Npad);
  };
  WT(W_rc1Wf, o_WfT,   256, 128, 256, 128);
  WT(W_rc1Wr, o_WrT,    32, 128,  32, 128);
  WT(W_init,  o_initT, 128, 128, 128, 128);
  WT(W_msg,   o_msgT,  128, 128, 128, 128);
  WT(W_sph,   o_sphT,  128,  24, 128,  32);
  WT(W_u1W1,  o_u1W1T, 128, 128, 128, 128);
  WT(W_u1W2,  o_u1W2T, 128, 128, 128, 128);
  WT(W_u2W1,  o_u2W1T,  24, 128,  32, 128);
  WT(W_u2W2,  o_u2W2T, 128, 128, 128, 128);
  WT(W_m1W1,  o_m1W1T, 128, 128, 128, 128);
  WT(W_m1W2,  o_m1W2T, 128, 128, 128, 128);
  WT(W_m1W3,  o_m1W3T, 128,   1, 128,  16);
  WT(W_qW1,   o_qW1T,  128, 256, 128, 256);
  WT(W_qW2,   o_qW2T,  256,   1, 256,  16);
  WT(W_u3W1,  o_u3W1T,  25, 128,  32, 128);
  WT(W_u3W2,  o_u3W2T, 128, 128, 128, 128);
  WT(W_m2W1,  o_m2W1T, 128, 128, 128, 128);
  WT(W_m2W2,  o_m2W2T, 128, 128, 128, 128);
  WT(W_m2W3,  o_m2W3T, 128,   1, 128,  16);

  mpnn_species<<<(NA*DFE + 255)/256, 256, 0, stream>>>(W_embed, Zi, amask, US(o_spbf));
  mpnn_ewald_prep<<<(NKV + 127)/128, 128, 0, stream>>>(cell, F(o_kt));

  mpnn_edges<<<NA/32, 64, 0, stream>>>(
      positions, cell, shifts, pmask, amask, others, US(o_spbf),
      US(o_WfT), b_rc1, US(o_WrT), US(o_msgT), US(o_sphT),
      F(o_upd), F(o_sphraw));

  mpnn_node1<<<NA/64, 128, 0, stream>>>(
      US(o_spbf), US(o_initT), b_init, F(o_upd),
      US(o_u1W1T), b_u1b1, US(o_u1W2T), b_u1b2,
      F(o_sphraw), W_td,
      US(o_u2W1T), b_u2b1, US(o_u2W2T), b_u2b2,
      US(o_m1W1T), b_m1b1, US(o_m1W2T), b_m1b2, US(o_m1W3T), b_m1b3,
      US(o_qW1T), b_qb1, US(o_qW2T), b_qb2,
      W_qs, amask,
      F(o_nodes), F(o_sph2), F(o_chg), F(o_e1));

  mpnn_sfact<<<NKV/4, 128, 0, stream>>>(positions, F(o_chg), F(o_kt), F(o_ScC), F(o_SsC));
  mpnn_pot<<<NA/256, 256, 0, stream>>>(positions, F(o_chg), F(o_kt), F(o_ScC), F(o_SsC),
                                       amask, F(o_potw));

  mpnn_node2<<<NA/64, 128, 0, stream>>>(
      F(o_nodes), F(o_sph2), F(o_potw), W_pot,
      US(o_u3W1T), b_u3b1, US(o_u3W2T), b_u3b2,
      US(o_m2W1T), b_m2b1, US(o_m2W2T), b_m2b2, US(o_m2W3T), b_m2b3,
      amask, F(o_e1), (float*)d_out);
}
